// CrossAttention_39298950758574
// MI455X (gfx1250) — matrix operand order repaired and checked by reading, not compiled, not run on hardware
//
#include <hip/hip_runtime.h>
#include <hip/hip_bf16.h>

typedef __attribute__((ext_vector_type(16))) __bf16 v16bf;
typedef __attribute__((ext_vector_type(8)))  __bf16 v8bf;
typedef __attribute__((ext_vector_type(8)))  float  v8f;
typedef __attribute__((ext_vector_type(4)))  unsigned u32x4;
typedef __attribute__((ext_vector_type(8)))  int      i32x8;
typedef __attribute__((ext_vector_type(4)))  int      i32x4;

#define B_     8
#define C_     512
#define N_     4096
#define M_     1024
#define CTX_   768
#define HEADS_ 8
#define DH_    64
#define INNER_ 512

#if __has_builtin(__builtin_amdgcn_tensor_load_to_lds) && __has_builtin(__builtin_amdgcn_s_wait_tensorcnt)
#define HAVE_TDM 1
#else
#define HAVE_TDM 0
#endif

__device__ __forceinline__ __bf16 f2bf(float f) { return (__bf16)f; }

// ---------------------------------------------------------------------------
// TDM: issue a 2D tile load (bf16 elements) global -> LDS.
//   lds_addr      : LDS byte offset of destination (contiguous tile_x*tile_y)
//   gaddr         : global byte address of tile start
//   tile_x/tile_y : tile extent (elements / rows)
//   stride_elems  : row stride of the global tensor in elements
// D# per CDNA5 ISA (count=1, type=2, data_size=2B,
// huge tensor dims so the in-bounds tile is never zero-filled by OOB logic).
// ---------------------------------------------------------------------------
#if HAVE_TDM
__device__ __forceinline__ void tdm_load_tile_bf16(unsigned lds_addr, const void* gaddr,
                                                   unsigned tile_x, unsigned tile_y,
                                                   unsigned stride_elems) {
  unsigned long long ga = (unsigned long long)gaddr;
  const unsigned dim0 = 0x40000000u, dim1 = 0x40000000u;
  u32x4 g0;
  g0[0] = 1u;                                             // count=1, user mode
  g0[1] = lds_addr;                                       // LDS byte address
  g0[2] = (unsigned)(ga & 0xffffffffu);                   // global_addr[31:0]
  g0[3] = (unsigned)((ga >> 32) & 0x01ffffffu) | (2u << 30); // addr[56:32] | type=2
  i32x8 g1;
  g1[0] = (int)(1u << 16);                                // data_size=1 (2 bytes)
  g1[1] = (int)((dim0 & 0xffffu) << 16);                  // tensor_dim0[15:0]
  g1[2] = (int)((dim0 >> 16) | ((dim1 & 0xffffu) << 16)); // dim0[31:16] | dim1[15:0]
  g1[3] = (int)((dim1 >> 16) | (tile_x << 16));           // dim1[31:16] | tile_dim0
  g1[4] = (int)tile_y;                                    // tile_dim1 (tile_dim2=0)
  g1[5] = (int)stride_elems;                              // tensor_dim0_stride[31:0]
  g1[6] = 0;                                              // stride hi / dim1_stride lo
  g1[7] = 0;
  i32x4 z4 = {0, 0, 0, 0};
#if defined(__clang_major__) && (__clang_major__ >= 23)
  i32x8 z8 = {0, 0, 0, 0, 0, 0, 0, 0};
  __builtin_amdgcn_tensor_load_to_lds(g0, g1, z4, z4, z8, 0);
#else
  __builtin_amdgcn_tensor_load_to_lds(g0, g1, z4, z4, 0);
#endif
}
#endif

// ---------------------------------------------------------------------------
// Transpose f32 [rows][cols] -> bf16 [cols][rows], batched via blockIdx.z
// ---------------------------------------------------------------------------
__global__ __launch_bounds__(256)
void k_transpose_cvt(const float* __restrict__ in, __bf16* __restrict__ out,
                     int rows, int cols, long inBatch, long outBatch) {
  __shared__ float tile[32][33];
  const float* ib = in + (long)blockIdx.z * inBatch;
  __bf16* ob = out + (long)blockIdx.z * outBatch;
  int x = blockIdx.x * 32 + threadIdx.x;
  int y0 = blockIdx.y * 32;
#pragma unroll
  for (int i = 0; i < 32; i += 8)
    tile[threadIdx.y + i][threadIdx.x] = ib[(long)(y0 + threadIdx.y + i) * cols + x];
  __syncthreads();
  int ox = y0 + threadIdx.x;
  int oy0 = blockIdx.x * 32;
#pragma unroll
  for (int i = 0; i < 32; i += 8)
    ob[(long)(oy0 + threadIdx.y + i) * rows + ox] = f2bf(tile[threadIdx.x][threadIdx.y + i]);
}

__global__ __launch_bounds__(256)
void k_cvt_bf16(const float* __restrict__ in, __bf16* __restrict__ out, long n) {
  long i = (long)blockIdx.x * blockDim.x + threadIdx.x;
  if (i < n) out[i] = f2bf(in[i]);
}

// ---------------------------------------------------------------------------
// Fragment helpers (wave32; C/D per ISA 7.12.2; A/B use contiguous k order, not the ISA table).
// A (16x32 bf16): lane holds row (lane&15); elems 0..15 = K 16*half+0..15
// B (32x16 bf16): lane holds col (lane&15); elems 0..15 = K 16*half+0..15
// C/D (16x16 f32): VGPR r = row r (lanes 0-15) / 8+r (lanes 16-31), col lane&15
// ---------------------------------------------------------------------------
__device__ __forceinline__ v16bf load_a_frag(const __bf16* rowPtr, int hl) {
  v8bf lo = *(const v8bf*)(rowPtr + 16 * hl);
  v8bf hi = *(const v8bf*)(rowPtr + 16 * hl + 8);
  v16bf a;
#pragma unroll
  for (int i = 0; i < 8; ++i) { a[i] = lo[i]; a[8 + i] = hi[i]; }
  return a;
}

// ---------------------------------------------------------------------------
// Projection GEMM: out = A[b] (Mrows x K) @ W^T (K x 512).
// Each wave: 32x64 tile (2 A-frags x 4 B-frags -> 8 WMMA per K step).
// vtrans==0: head-split [B][H][Mrows][64]; vtrans==1: transposed [B][H][64][Mrows]
// ---------------------------------------------------------------------------
__global__ __launch_bounds__(128)
void k_gemm_proj(const __bf16* __restrict__ A, const __bf16* __restrict__ BT,
                 __bf16* __restrict__ out, int Mrows, int K, int vtrans) {
  const int wave = threadIdx.x >> 5, lane = threadIdx.x & 31;
  const int hl = lane >> 4, l16 = lane & 15;
  const int b = blockIdx.z;
  const int row0 = blockIdx.y * 128 + wave * 32;
  const int col0 = blockIdx.x * 64;
  const __bf16* Ab = A + (long)b * Mrows * K;
  v8f acc[2][4] = {};
  for (int k0 = 0; k0 < K; k0 += 32) {
    v16bf a0 = load_a_frag(Ab + (long)(row0 + l16) * K + k0, hl);
    v16bf a1 = load_a_frag(Ab + (long)(row0 + 16 + l16) * K + k0, hl);
#pragma unroll
    for (int j = 0; j < 4; ++j) {
      v16bf bb = *(const v16bf*)(BT + (long)(col0 + 16 * j + l16) * K + k0 + 16 * hl);
      acc[0][j] = __builtin_amdgcn_wmma_f32_16x16x32_bf16(false, a0, false, bb,
                                                          (short)0, acc[0][j], false, false);
      acc[1][j] = __builtin_amdgcn_wmma_f32_16x16x32_bf16(false, a1, false, bb,
                                                          (short)0, acc[1][j], false, false);
    }
  }
#pragma unroll
  for (int ia = 0; ia < 2; ++ia) {
#pragma unroll
    for (int j = 0; j < 4; ++j) {
      int i = col0 + 16 * j + l16;
      int h = i >> 6, d = i & 63;
#pragma unroll
      for (int r = 0; r < 8; ++r) {
        int m = row0 + 16 * ia + (hl ? 8 : 0) + r;
        long idx = vtrans ? (((long)b * HEADS_ + h) * DH_ + d) * Mrows + m
                          : (((long)b * HEADS_ + h) * Mrows + m) * DH_ + d;
        out[idx] = f2bf(acc[ia][j][r]);
      }
    }
  }
}

// ---------------------------------------------------------------------------
// Flash attention. Block = 4 waves x 16 Q rows of one (b, head).
// K/V 64-wide M-chunks staged into LDS by the Tensor Data Mover (wave 0,
// double-buffered, s_wait_tensorcnt + barrier); all waves consume from LDS.
// Per chunk: 8 WMMA for S, online softmax, P through LDS, 8 WMMA for P*V.
// ---------------------------------------------------------------------------
__global__ __launch_bounds__(128)
void k_attn(const __bf16* __restrict__ q,   // [B][H][N][64]
            const __bf16* __restrict__ kk,  // [B][H][M][64]
            const __bf16* __restrict__ vT,  // [B][H][64][M]
            __bf16* __restrict__ aout) {    // [B][N][512]
  const int wave = threadIdx.x >> 5, lane = threadIdx.x & 31;
  const int hl = lane >> 4, l16 = lane & 15;
  const int b = blockIdx.z, hd = blockIdx.y;
  const int n0 = blockIdx.x * 64 + wave * 16;
  const long bh = (long)b * HEADS_ + hd;

  __shared__ __bf16 ktile[2][64][64];   // K rows (m) x d
  __shared__ __bf16 vtile[2][64][64];   // V^T rows (d) x m
  __shared__ __bf16 plds[4][16][72];    // per-wave P tile, 16B-aligned rows

  const __bf16* kb = kk + bh * M_ * DH_;
  const __bf16* vb = vT + bh * DH_ * M_;

  const __bf16* qr = q + (bh * N_ + n0 + l16) * DH_;
  v16bf qa[2];
#pragma unroll
  for (int kt = 0; kt < 2; ++kt) qa[kt] = load_a_frag(qr + kt * 32, hl);

  const float sc = 0.125f * 1.44269504088896340736f;  // dim^-0.5 * log2(e)
  float rm[8], rs[8];
  v8f o[4] = {};
#pragma unroll
  for (int r = 0; r < 8; ++r) { rm[r] = -3.0e38f; rs[r] = 0.f; }

#if HAVE_TDM
  if (wave == 0) {
    tdm_load_tile_bf16((unsigned)(unsigned long long)&ktile[0][0][0], kb, 64, 64, DH_);
    tdm_load_tile_bf16((unsigned)(unsigned long long)&vtile[0][0][0], vb, 64, 64, M_);
    __builtin_amdgcn_s_wait_tensorcnt(0);
  }
  __syncthreads();
#endif

  for (int m0 = 0; m0 < M_; m0 += 64) {
#if HAVE_TDM
    const int cur = (m0 >> 6) & 1, nxt = cur ^ 1;
    if (wave == 0 && (m0 + 64) < M_) {   // prefetch next chunk while computing
      tdm_load_tile_bf16((unsigned)(unsigned long long)&ktile[nxt][0][0],
                         kb + (long)(m0 + 64) * DH_, 64, 64, DH_);
      tdm_load_tile_bf16((unsigned)(unsigned long long)&vtile[nxt][0][0],
                         vb + (m0 + 64), 64, 64, M_);
    }
#else
    const int cur = 0;
    for (int i = threadIdx.x; i < (64 * 64) / 8; i += 128) {  // cooperative copy
      int row = i >> 3, c8 = (i & 7) * 8;
      *(v8bf*)&ktile[0][row][c8] = *(const v8bf*)(kb + (long)(m0 + row) * DH_ + c8);
      *(v8bf*)&vtile[0][row][c8] = *(const v8bf*)(vb + (long)row * M_ + m0 + c8);
    }
    __syncthreads();
#endif

    // ---- S = Q K^T : four 16x16 column fragments of the 16x64 score chunk
    v8f s[4];
#pragma unroll
    for (int j = 0; j < 4; ++j) {
      v8f sj = {};
#pragma unroll
      for (int kt = 0; kt < 2; ++kt) {
        v16bf bbb = *(const v16bf*)(&ktile[cur][16 * j + l16][32 * kt + 16 * hl]);
        sj = __builtin_amdgcn_wmma_f32_16x16x32_bf16(false, qa[kt], false, bbb,
                                                     (short)0, sj, false, false);
      }
      s[j] = sj;
    }

    // ---- online softmax (row cols live in one 16-lane half)
#pragma unroll
    for (int r = 0; r < 8; ++r) {
      float a0 = s[0][r] * sc, a1 = s[1][r] * sc, a2 = s[2][r] * sc, a3 = s[3][r] * sc;
      float t = fmaxf(fmaxf(a0, a1), fmaxf(a2, a3));
#pragma unroll
      for (int msk = 1; msk <= 8; msk <<= 1) t = fmaxf(t, __shfl_xor(t, msk, 32));
      float nm = fmaxf(rm[r], t);
      float corr = exp2f(rm[r] - nm);
      rm[r] = nm;
      float p0 = exp2f(a0 - nm), p1 = exp2f(a1 - nm);
      float p2 = exp2f(a2 - nm), p3 = exp2f(a3 - nm);
      rs[r] = rs[r] * corr + ((p0 + p1) + (p2 + p3));
#pragma unroll
      for (int jd = 0; jd < 4; ++jd) o[jd][r] *= corr;
      int row = (hl ? 8 : 0) + r;
      plds[wave][row][l16] = f2bf(p0);
      plds[wave][row][16 + l16] = f2bf(p1);
      plds[wave][row][32 + l16] = f2bf(p2);
      plds[wave][row][48 + l16] = f2bf(p3);
    }
    asm volatile("s_wait_dscnt 0" ::: "memory");  // P writes -> A-layout reads

    v16bf pa[2];
#pragma unroll
    for (int kt = 0; kt < 2; ++kt) {
      v8bf lo = *(const v8bf*)(&plds[wave][l16][32 * kt + 16 * hl]);
      v8bf hi = *(const v8bf*)(&plds[wave][l16][32 * kt + 16 * hl + 8]);
#pragma unroll
      for (int i = 0; i < 8; ++i) { pa[kt][i] = lo[i]; pa[kt][8 + i] = hi[i]; }
    }

    // ---- O += P V
#pragma unroll
    for (int kt = 0; kt < 2; ++kt) {
#pragma unroll
      for (int jd = 0; jd < 4; ++jd) {
        v16bf vv = *(const v16bf*)(&vtile[cur][16 * jd + l16][32 * kt + 16 * hl]);
        o[jd] = __builtin_amdgcn_wmma_f32_16x16x32_bf16(false, pa[kt], false, vv,
                                                        (short)0, o[jd], false, false);
      }
    }

#if HAVE_TDM
    if (wave == 0) __builtin_amdgcn_s_wait_tensorcnt(0);
    __syncthreads();   // next buffer ready; this buffer free for overwrite
#else
    __syncthreads();
#endif
  }

  // ---- normalize + store
#pragma unroll
  for (int r = 0; r < 8; ++r) {
    float t = rs[r];
#pragma unroll
    for (int msk = 1; msk <= 8; msk <<= 1) t += __shfl_xor(t, msk, 32);
    float inv = 1.0f / t;
    int n = n0 + (hl ? 8 : 0) + r;
#pragma unroll
    for (int jd = 0; jd < 4; ++jd) {
      int d = 16 * jd + l16;
      aout[((long)b * N_ + n) * INNER_ + hd * DH_ + d] = f2bf(o[jd][r] * inv);
    }
  }
}

// ---------------------------------------------------------------------------
// O-projection + bias + residual: out[b,c,n] = attn[b,n,:]@Wo + bo[c] + x[b,c,n]
// Wave tile 32x64 like k_gemm_proj.
// ---------------------------------------------------------------------------
__global__ __launch_bounds__(128)
void k_gemm_out(const __bf16* __restrict__ A,    // [B][N][512]
                const __bf16* __restrict__ WoT,  // [512][512] (out,in)
                const float* __restrict__ bo,
                const float* __restrict__ x,     // [B][512][4096]
                float* __restrict__ out) {
  const int wave = threadIdx.x >> 5, lane = threadIdx.x & 31;
  const int hl = lane >> 4, l16 = lane & 15;
  const int b = blockIdx.z;
  const int row0 = blockIdx.y * 128 + wave * 32;  // n
  const int col0 = blockIdx.x * 64;               // c
  const __bf16* Ab = A + (long)b * N_ * INNER_;
  v8f acc[2][4] = {};
  for (int k0 = 0; k0 < INNER_; k0 += 32) {
    v16bf a0 = load_a_frag(Ab + (long)(row0 + l16) * INNER_ + k0, hl);
    v16bf a1 = load_a_frag(Ab + (long)(row0 + 16 + l16) * INNER_ + k0, hl);
#pragma unroll
    for (int j = 0; j < 4; ++j) {
      v16bf bb = *(const v16bf*)(WoT + (long)(col0 + 16 * j + l16) * INNER_ + k0 + 16 * hl);
      acc[0][j] = __builtin_amdgcn_wmma_f32_16x16x32_bf16(false, a0, false, bb,
                                                          (short)0, acc[0][j], false, false);
      acc[1][j] = __builtin_amdgcn_wmma_f32_16x16x32_bf16(false, a1, false, bb,
                                                          (short)0, acc[1][j], false, false);
    }
  }
#pragma unroll
  for (int ia = 0; ia < 2; ++ia) {
#pragma unroll
    for (int j = 0; j < 4; ++j) {
      int c = col0 + 16 * j + l16;
      float bias = bo[c];
#pragma unroll
      for (int r = 0; r < 8; ++r) {   // 8 consecutive n per lane -> coalesced
        int n = row0 + 16 * ia + (hl ? 8 : 0) + r;
        long idx = ((long)b * C_ + c) * N_ + n;
        out[idx] = acc[ia][j][r] + bias + x[idx];
      }
    }
  }
}

// ---------------------------------------------------------------------------
extern "C" void kernel_launch(void* const* d_in, const int* in_sizes, int n_in,
                              void* d_out, int out_size, void* d_ws, size_t ws_size,
                              hipStream_t stream) {
  (void)in_sizes; (void)n_in; (void)out_size; (void)ws_size;
  const float* x   = (const float*)d_in[0];
  const float* ctx = (const float*)d_in[1];
  const float* Wq  = (const float*)d_in[2];
  const float* Wk  = (const float*)d_in[3];
  const float* Wv  = (const float*)d_in[4];
  const float* Wo  = (const float*)d_in[5];
  const float* bo  = (const float*)d_in[6];
  float* out = (float*)d_out;

  char* ws = (char*)d_ws;
  size_t off = 0;
  auto alloc = [&](size_t bytes) {
    char* p = ws + off;
    off += (bytes + 255) & ~((size_t)255);
    return p;
  };
  __bf16* xT   = (__bf16*)alloc((size_t)B_ * N_ * C_ * 2);        // [B][N][C]
  __bf16* ctxb = (__bf16*)alloc((size_t)B_ * M_ * CTX_ * 2);      // [B][M][CTX]
  __bf16* WqT  = (__bf16*)alloc((size_t)C_ * INNER_ * 2);
  __bf16* WkT  = (__bf16*)alloc((size_t)CTX_ * INNER_ * 2);
  __bf16* WvT  = (__bf16*)alloc((size_t)CTX_ * INNER_ * 2);
  __bf16* WoT  = (__bf16*)alloc((size_t)INNER_ * C_ * 2);
  __bf16* qb   = (__bf16*)alloc((size_t)B_ * HEADS_ * N_ * DH_ * 2);
  __bf16* kbuf = (__bf16*)alloc((size_t)B_ * HEADS_ * M_ * DH_ * 2);
  __bf16* vTb  = (__bf16*)alloc((size_t)B_ * HEADS_ * DH_ * M_ * 2);
  __bf16* ao   = (__bf16*)alloc((size_t)B_ * N_ * INNER_ * 2);

  dim3 tb(32, 8);
  k_transpose_cvt<<<dim3(N_ / 32, C_ / 32, B_), tb, 0, stream>>>(
      x, xT, C_, N_, (long)C_ * N_, (long)N_ * C_);
  k_transpose_cvt<<<dim3(INNER_ / 32, C_ / 32, 1), tb, 0, stream>>>(Wq, WqT, C_, INNER_, 0, 0);
  k_transpose_cvt<<<dim3(INNER_ / 32, CTX_ / 32, 1), tb, 0, stream>>>(Wk, WkT, CTX_, INNER_, 0, 0);
  k_transpose_cvt<<<dim3(INNER_ / 32, CTX_ / 32, 1), tb, 0, stream>>>(Wv, WvT, CTX_, INNER_, 0, 0);
  k_transpose_cvt<<<dim3(C_ / 32, INNER_ / 32, 1), tb, 0, stream>>>(Wo, WoT, INNER_, C_, 0, 0);
  long nctx = (long)B_ * M_ * CTX_;
  k_cvt_bf16<<<dim3((unsigned)(nctx / 256)), dim3(256), 0, stream>>>(ctx, ctxb, nctx);

  k_gemm_proj<<<dim3(INNER_ / 64, N_ / 128, B_), dim3(128), 0, stream>>>(xT, WqT, qb, N_, C_, 0);
  k_gemm_proj<<<dim3(INNER_ / 64, M_ / 128, B_), dim3(128), 0, stream>>>(ctxb, WkT, kbuf, M_, CTX_, 0);
  k_gemm_proj<<<dim3(INNER_ / 64, M_ / 128, B_), dim3(128), 0, stream>>>(ctxb, WvT, vTb, M_, CTX_, 1);

  k_attn<<<dim3(N_ / 64, HEADS_, B_), dim3(128), 0, stream>>>(qb, kbuf, vTb, ao);

  k_gemm_out<<<dim3(C_ / 64, N_ / 128, B_), dim3(128), 0, stream>>>(ao, WoT, bo, x, out);
}
